// IA3SelfAttention_1692217115016
// MI455X (gfx1250) — matrix operand order repaired and checked by reading, not compiled, not run on hardware
//
#include <hip/hip_runtime.h>
#include <math.h>

#define BQ 4
#define SQ 4096
#define HD 768

typedef __attribute__((ext_vector_type(16))) __bf16 v16bf;
typedef __attribute__((ext_vector_type(8)))  __bf16 v8bf;
typedef __attribute__((ext_vector_type(4)))  __bf16 v4bf;
typedef __attribute__((ext_vector_type(8)))  float  v8f;

// Build a 16x32 bf16 A-fragment: lane m = lane&15 holds
// K [koff..koff+15] contiguous with koff = 16*(lane>>4).
static __device__ __forceinline__ v16bf make_afrag(const __bf16* p) {
  v8bf lo = *(const v8bf*)p;
  v8bf hi = *(const v8bf*)(p + 8);
  v16bf a;
#pragma unroll
  for (int i = 0; i < 8; ++i) { a[i] = lo[i]; a[i + 8] = hi[i]; }
  return a;
}

// ---------------------------------------------------------------------------
// Prep: x fp32 -> bf16
// ---------------------------------------------------------------------------
__global__ void cvt_x_kernel(const float* __restrict__ src,
                             __bf16* __restrict__ dst, int n4) {
  int i = blockIdx.x * blockDim.x + threadIdx.x;
  if (i < n4) {
    float4 v = ((const float4*)src)[i];
    v4bf o;
    o[0] = (__bf16)v.x; o[1] = (__bf16)v.y;
    o[2] = (__bf16)v.z; o[3] = (__bf16)v.w;
    *(v4bf*)&dst[(size_t)i * 4] = o;
  }
}

// Prep: W fp32 -> bf16 with IA3 scaler folded into rows; bias scaled too.
__global__ void prep_w_kernel(const float* __restrict__ W,
                              const float* __restrict__ scale,
                              const float* __restrict__ bias,
                              __bf16* __restrict__ Wb,
                              float* __restrict__ bs) {
  int idx = blockIdx.x * blockDim.x + threadIdx.x;
  if (idx < HD * HD) {
    int o = idx / HD;
    float sc = scale ? scale[o] : 1.0f;
    Wb[idx] = (__bf16)(W[idx] * sc);
  }
  if (idx < HD) {
    float sc = scale ? scale[idx] : 1.0f;
    bs[idx] = bias[idx] * sc;
  }
}

// ---------------------------------------------------------------------------
// QKV projection: out[s,o] = sum_h X[s,h]*W[o,h] + b[o].
// Each wave computes a 16(M) x 64(N) tile; K = 768 in 24 WMMA steps.
// transposed==1 stores out as [B][H][S] (for the PV-GEMM B operand).
// ---------------------------------------------------------------------------
__global__ __launch_bounds__(256) void qkv_gemm_kernel(
    const __bf16* __restrict__ X, const __bf16* __restrict__ W,
    const float* __restrict__ bias, __bf16* __restrict__ out, int transposed) {
  const int tid  = threadIdx.x;
  const int wave = tid >> 5;
  const int lane = tid & 31;
  const int m16  = lane & 15;
  const int hi   = lane >> 4;
  const int gid  = blockIdx.x * 8 + wave;   // wave-tile id
  const int mt   = gid / 12;                // 1024 M tiles (B*S/16)
  const int nt   = gid % 12;                // 12 N groups of 64

  v8f acc[4];
#pragma unroll
  for (int j = 0; j < 4; ++j)
#pragma unroll
    for (int i = 0; i < 8; ++i) acc[j][i] = 0.0f;

  const __bf16* xrow = X + (size_t)(mt * 16 + m16) * HD;
#pragma unroll 4
  for (int kk = 0; kk < 24; ++kk) {
    v16bf a = make_afrag(&xrow[kk * 32 + hi * 16]);
#pragma unroll
    for (int j = 0; j < 4; ++j) {
      const int o = nt * 64 + j * 16 + m16;            // N lane = output chan
      v16bf bb = *(const v16bf*)&W[(size_t)o * HD + kk * 32 + hi * 16];
      acc[j] = __builtin_amdgcn_wmma_f32_16x16x32_bf16(
          false, a, false, bb, (short)0, acc[j], false, false);
    }
  }

#pragma unroll
  for (int j = 0; j < 4; ++j) {
    const int o  = nt * 64 + j * 16 + m16;
    const float bj = bias[o];
    if (!transposed) {
#pragma unroll
      for (int i = 0; i < 8; ++i)
        out[(size_t)(mt * 16 + i + 8 * hi) * HD + o] = (__bf16)(acc[j][i] + bj);
    } else {
      const int row0 = mt * 16;
      const int bb   = row0 >> 12;                 // batch (S=4096 rows)
      const int s0   = (row0 & 4095) + 8 * hi;
      v4bf* dst = (v4bf*)&out[((size_t)bb * HD + o) * SQ + s0];
      v8bf pv;
#pragma unroll
      for (int i = 0; i < 8; ++i) pv[i] = (__bf16)(acc[j][i] + bj);
      *(v8bf*)dst = pv;                            // 16B contiguous store
    }
  }
}

// ---------------------------------------------------------------------------
// Flash attention: one workgroup (8 waves) per (batch, 16-query block).
//  Phase A : wave w -> 16x16 score tile at keys [kv0+16w, +16)   (24 WMMAs)
//  softmax : online max/sum across the 128-wide tile via shfl + LDS
//  Phase B : wave w -> 96 output columns [96w, 96w+96)           (24 WMMAs)
// ---------------------------------------------------------------------------
__global__ __launch_bounds__(256) void ia3_attn_kernel(
    const __bf16* __restrict__ Q, const __bf16* __restrict__ K,
    const __bf16* __restrict__ VT, float* __restrict__ out) {
  __shared__ __align__(16) __bf16 q_s[16 * 776];  // 768 + 8 pad (banking)
  __shared__ __align__(16) __bf16 p_s[16 * 136];  // 128 + 8 pad
  __shared__ float red_max[8 * 16];
  __shared__ float red_sum[8 * 16];
  __shared__ float m_s[16], l_s[16], alpha_s[16];

  const int tid  = threadIdx.x;
  const int wave = tid >> 5;
  const int lane = tid & 31;
  const int m16  = lane & 15;
  const int hi   = lane >> 4;
  const int b    = blockIdx.y;
  const int q0   = blockIdx.x * 16;

  // Stage the 16x768 bf16 Q block into LDS (cooperative, 16B chunks).
  const size_t qbase = ((size_t)b * SQ + q0) * HD;
  for (int i = tid; i < 16 * 96; i += 256) {
    int r = i / 96, c = (i % 96) * 8;
    *(uint4*)&q_s[r * 776 + c] = *(const uint4*)&Q[qbase + (size_t)r * HD + c];
  }
  if (tid < 16) { m_s[tid] = -INFINITY; l_s[tid] = 0.0f; }

  v8f oacc[6];
#pragma unroll
  for (int t = 0; t < 6; ++t)
#pragma unroll
    for (int i = 0; i < 8; ++i) oacc[t][i] = 0.0f;

  __syncthreads();

  for (int kv0 = 0; kv0 < SQ; kv0 += 128) {
    // ---------------- Phase A: scores ----------------
    const __bf16* krow = K + ((size_t)b * SQ + kv0 + wave * 16 + m16) * HD;
    if (kv0 + 128 < SQ) __builtin_prefetch(krow + (size_t)128 * HD, 0, 1);

    v8f sacc;
#pragma unroll
    for (int i = 0; i < 8; ++i) sacc[i] = 0.0f;
#pragma unroll 4
    for (int kk = 0; kk < 24; ++kk) {
      v16bf a  = make_afrag(&q_s[m16 * 776 + kk * 32 + hi * 16]);
      v16bf bb = *(const v16bf*)&krow[kk * 32 + hi * 16];
      sacc = __builtin_amdgcn_wmma_f32_16x16x32_bf16(
          false, a, false, bb, (short)0, sacc, false, false);
    }

    // partial row-max: element i is row (i + 8*hi), row spread over 16 lanes
    float pm[8];
#pragma unroll
    for (int i = 0; i < 8; ++i) pm[i] = sacc[i];
#pragma unroll
    for (int off = 8; off >= 1; off >>= 1)
#pragma unroll
      for (int i = 0; i < 8; ++i)
        pm[i] = fmaxf(pm[i], __shfl_xor(pm[i], off, 16));
    if (m16 == 0) {
#pragma unroll
      for (int i = 0; i < 8; ++i) red_max[wave * 16 + i + 8 * hi] = pm[i];
    }
    __syncthreads();

    if (tid < 16) {
      float bm = red_max[tid];
#pragma unroll
      for (int w = 1; w < 8; ++w) bm = fmaxf(bm, red_max[w * 16 + tid]);
      float mo = m_s[tid];
      float mn = fmaxf(mo, bm);
      float al = __expf(mo - mn);
      m_s[tid] = mn; alpha_s[tid] = al; l_s[tid] *= al;
    }
    __syncthreads();

    // ---------------- Phase A2: P = exp(S - m), row sums ----------------
    float ps[8];
#pragma unroll
    for (int i = 0; i < 8; ++i) ps[i] = __expf(sacc[i] - m_s[i + 8 * hi]);
#pragma unroll
    for (int i = 0; i < 8; ++i)
      p_s[(i + 8 * hi) * 136 + wave * 16 + m16] = (__bf16)ps[i];

    float sm[8];
#pragma unroll
    for (int i = 0; i < 8; ++i) sm[i] = ps[i];
#pragma unroll
    for (int off = 8; off >= 1; off >>= 1)
#pragma unroll
      for (int i = 0; i < 8; ++i) sm[i] += __shfl_xor(sm[i], off, 16);
    if (m16 == 0) {
#pragma unroll
      for (int i = 0; i < 8; ++i) red_sum[wave * 16 + i + 8 * hi] = sm[i];
    }
    __syncthreads();

    if (tid < 16) {
      float s = 0.0f;
#pragma unroll
      for (int w = 0; w < 8; ++w) s += red_sum[w * 16 + tid];
      l_s[tid] += s;
    }

    // ---------------- Phase B: O = alpha*O + P @ V ----------------
    float arow[8];
#pragma unroll
    for (int i = 0; i < 8; ++i) arow[i] = alpha_s[i + 8 * hi];

    v16bf pa[4];
#pragma unroll
    for (int k2 = 0; k2 < 4; ++k2)
      pa[k2] = make_afrag(&p_s[m16 * 136 + k2 * 32 + hi * 16]);

    const __bf16* vbase = VT + ((size_t)b * HD + wave * 96 + m16) * SQ + kv0;
    if (kv0 + 128 < SQ) __builtin_prefetch(vbase + 128, 0, 1);

#pragma unroll
    for (int t = 0; t < 6; ++t) {
      v8f oc = oacc[t];
#pragma unroll
      for (int i = 0; i < 8; ++i) oc[i] *= arow[i];
#pragma unroll
      for (int k2 = 0; k2 < 4; ++k2) {
        v16bf vb =
            *(const v16bf*)&vbase[(size_t)t * 16 * SQ + k2 * 32 + hi * 16];
        oc = __builtin_amdgcn_wmma_f32_16x16x32_bf16(
            false, pa[k2], false, vb, (short)0, oc, false, false);
      }
      oacc[t] = oc;
    }
    __syncthreads();  // p_s / stats stable for next iteration; l_s settled
  }

  // ---------------- Epilogue: O / l ----------------
  float inv[8];
#pragma unroll
  for (int i = 0; i < 8; ++i) inv[i] = 1.0f / l_s[i + 8 * hi];
#pragma unroll
  for (int t = 0; t < 6; ++t) {
    const int h = wave * 96 + t * 16 + m16;
#pragma unroll
    for (int i = 0; i < 8; ++i) {
      const int r = q0 + i + 8 * hi;
      out[((size_t)b * SQ + r) * HD + h] = oacc[t][i] * inv[i];
    }
  }
}

// ---------------------------------------------------------------------------
extern "C" void kernel_launch(void* const* d_in, const int* in_sizes, int n_in,
                              void* d_out, int out_size, void* d_ws,
                              size_t ws_size, hipStream_t stream) {
  (void)in_sizes; (void)n_in; (void)out_size; (void)ws_size;
  const float* x  = (const float*)d_in[0];
  const float* Wq = (const float*)d_in[1];
  const float* bq = (const float*)d_in[2];
  const float* Wk = (const float*)d_in[3];
  const float* bk = (const float*)d_in[4];
  const float* Wv = (const float*)d_in[5];
  const float* bv = (const float*)d_in[6];
  const float* lk = (const float*)d_in[7];
  const float* lv = (const float*)d_in[8];

  char* ws = (char*)d_ws;
  size_t off = 0;
  auto walloc = [&](size_t bytes) -> void* {
    void* p = ws + off;
    off = (off + bytes + 255) & ~(size_t)255;
    return p;
  };
  const size_t XSZ = (size_t)BQ * SQ * HD;  // 12.58M elements
  __bf16* Xb  = (__bf16*)walloc(XSZ * 2);
  __bf16* Qb  = (__bf16*)walloc(XSZ * 2);
  __bf16* Kb  = (__bf16*)walloc(XSZ * 2);
  __bf16* VTb = (__bf16*)walloc(XSZ * 2);           // [B][H][S]
  __bf16* Wqb = (__bf16*)walloc((size_t)HD * HD * 2);
  __bf16* Wkb = (__bf16*)walloc((size_t)HD * HD * 2);
  __bf16* Wvb = (__bf16*)walloc((size_t)HD * HD * 2);
  float*  bqs = (float*)walloc(HD * 4);
  float*  bks = (float*)walloc(HD * 4);
  float*  bvs = (float*)walloc(HD * 4);

  const int n4 = (int)(XSZ / 4);
  cvt_x_kernel<<<(n4 + 255) / 256, 256, 0, stream>>>(x, Xb, n4);
  const int wgrid = (HD * HD + 255) / 256;
  prep_w_kernel<<<wgrid, 256, 0, stream>>>(Wq, nullptr, bq, Wqb, bqs);
  prep_w_kernel<<<wgrid, 256, 0, stream>>>(Wk, lk,      bk, Wkb, bks);
  prep_w_kernel<<<wgrid, 256, 0, stream>>>(Wv, lv,      bv, Wvb, bvs);

  const int tiles = (BQ * SQ / 16) * (HD / 64);  // 12288 wave tiles
  qkv_gemm_kernel<<<tiles / 8, 256, 0, stream>>>(Xb, Wqb, bqs, Qb, 0);
  qkv_gemm_kernel<<<tiles / 8, 256, 0, stream>>>(Xb, Wkb, bks, Kb, 0);
  qkv_gemm_kernel<<<tiles / 8, 256, 0, stream>>>(Xb, Wvb, bvs, VTb, 1);

  dim3 agrid(SQ / 16, BQ);
  ia3_attn_kernel<<<agrid, 256, 0, stream>>>(Qb, Kb, VTb, (float*)d_out);
}
